// InferLinearQuantizer_30459908063608
// MI455X (gfx1250) — 8-bit matrix operand order repaired and checked by reading, compiled, not run on hardware
//
#include <hip/hip_runtime.h>

typedef __attribute__((ext_vector_type(8))) int   v8i;
typedef __attribute__((ext_vector_type(4))) int   v4i;
typedef __attribute__((ext_vector_type(2))) int   v2i;
typedef __attribute__((ext_vector_type(4))) float v4f;

#define N_TOKENS 8192
#define IN_F     4096
#define OUT_F    4096
#define BM 128
#define BN 128
#define BK 64
#define LDK 80
#define KSTEPS (IN_F / BK)

#if defined(__AMDGCN__) && __has_builtin(__builtin_amdgcn_global_load_async_to_lds_b128)
#define HAVE_ASYNC_LDS 1
typedef __attribute__((address_space(1))) v4i gv4i;
typedef __attribute__((address_space(3))) v4i lv4i;
#else
#define HAVE_ASYNC_LDS 0
#endif

__device__ __forceinline__ void copy16_g2l(const signed char* g, signed char* l) {
#if HAVE_ASYNC_LDS
    __builtin_amdgcn_global_load_async_to_lds_b128((gv4i*)g, (lv4i*)l, 0, 0);
#else
    *(v4i*)l = *(const v4i*)g;
#endif
}

__device__ __forceinline__ void wait_stage() {
#if HAVE_ASYNC_LDS
#if __has_builtin(__builtin_amdgcn_s_wait_asynccnt)
    __builtin_amdgcn_s_wait_asynccnt(0);
#else
    asm volatile("s_wait_asynccnt 0" ::: "memory");
#endif
#endif
}

__device__ __forceinline__ unsigned nib_lo(unsigned p) {
    return (p & 0xFu) | (((p >> 4) & 0xFu) << 8) |
           (((p >> 8) & 0xFu) << 16) | (((p >> 12) & 0xFu) << 24);
}
__device__ __forceinline__ unsigned nib_hi(unsigned p) {
    return ((p >> 16) & 0xFu) | (((p >> 20) & 0xFu) << 8) |
           (((p >> 24) & 0xFu) << 16) | ((p >> 28) << 24);
}

__global__ __launch_bounds__(256) void quant_rows_kernel(
    const float* __restrict__ x, const float* __restrict__ act_scale,
    signed char* __restrict__ xq, int* __restrict__ rowsum)
{
    const int row = blockIdx.x;
    const int t   = threadIdx.x;
    const float inv = 1.0f / act_scale[0];
    const float* xr = x + (size_t)row * IN_F;
    const int base = t * 16;

    int acc = 0;
    int packed[4];
#pragma unroll
    for (int g = 0; g < 4; ++g) {
        v4f v = *(const v4f*)(xr + base + g * 4);
        int q0 = (int)fminf(127.f, fmaxf(-128.f, rintf(v.x * inv)));
        int q1 = (int)fminf(127.f, fmaxf(-128.f, rintf(v.y * inv)));
        int q2 = (int)fminf(127.f, fmaxf(-128.f, rintf(v.z * inv)));
        int q3 = (int)fminf(127.f, fmaxf(-128.f, rintf(v.w * inv)));
        acc += q0 + q1 + q2 + q3;
        packed[g] = (q0 & 255) | ((q1 & 255) << 8) | ((q2 & 255) << 16) | ((q3 & 255) << 24);
    }
    *(v4i*)(xq + (size_t)row * IN_F + base) = *(const v4i*)packed;

    __shared__ int red[256];
    red[t] = acc;
    __syncthreads();
#pragma unroll
    for (int off = 128; off > 0; off >>= 1) {
        if (t < off) red[t] += red[t + off];
        __syncthreads();
    }
    if (t == 0) rowsum[row] = red[0];
}

__global__ __launch_bounds__(256) void unpack_w_kernel(
    const int* __restrict__ pw, signed char* __restrict__ wq)
{
    const size_t w0 = ((size_t)blockIdx.x * 256 + threadIdx.x) * 2;
    const unsigned p0 = (unsigned)pw[w0];
    const unsigned p1 = (unsigned)pw[w0 + 1];
    v4i o;
    o[0] = (int)nib_lo(p0); o[1] = (int)nib_hi(p0);
    o[2] = (int)nib_lo(p1); o[3] = (int)nib_hi(p1);
    *(v4i*)(wq + w0 * 8) = o;
}

__device__ __forceinline__ v8i ld_fragA(const signed char* p) {
    v4i d0 = *(const v4i*)(p +  0);
    v4i d1 = *(const v4i*)(p + 32);
    v8i f;
    f[0] = d0[0]; f[1] = d0[1]; f[2] = d0[2]; f[3] = d0[3];
    f[4] = d1[0]; f[5] = d1[1]; f[6] = d1[2]; f[7] = d1[3];
    return f;
}

__device__ __forceinline__ v8i ld_fragB(const signed char* p) {
    v4i d0 = *(const v4i*)(p +  0);
    v4i d1 = *(const v4i*)(p + 32);
    v8i f;
    f[0] = d0[0]; f[1] = d0[1]; f[2] = d0[2]; f[3] = d0[3];
    f[4] = d1[0]; f[5] = d1[1]; f[6] = d1[2]; f[7] = d1[3];
    return f;
}

template <bool PRE>
__global__ __launch_bounds__(256) void w4a8_gemm_kernel(
    const signed char* __restrict__ xq,
    const signed char* __restrict__ wq,
    const int* __restrict__ packed_w,
    const int* __restrict__ rowsum,
    const float* __restrict__ scale,
    const float* __restrict__ zero_point,
    const float* __restrict__ bias,
    const float* __restrict__ act_scale,
    float* __restrict__ out)
{
    __shared__ __align__(16) signed char As[2][BM * LDK];
    __shared__ __align__(16) signed char Bs[2][BN * LDK];

    const int tid  = threadIdx.x;
    const int lane = tid & 31;
    const int wave = tid >> 5;
    const int wm   = wave >> 2;
    const int wn   = wave & 3;
    const int tile_m = blockIdx.x * BM;
    const int tile_o = blockIdx.y * BN;

    const int rlo = lane & 15;
    const int khA = (lane >> 4) << 4;
    const int khB = (lane >> 4) << 4;

    v8i acc[4][2] = {};

    auto stage = [&](int ks, int buf) {
#pragma unroll
        for (int it = 0; it < 2; ++it) {
            int c   = it * 256 + tid;
            int r   = c >> 2;
            int off = (c & 3) << 4;
            copy16_g2l(xq + (size_t)(tile_m + r) * IN_F + ks * BK + off,
                       &As[buf][r * LDK + off]);
        }
        if constexpr (PRE) {
#pragma unroll
            for (int it = 0; it < 2; ++it) {
                int c   = it * 256 + tid;
                int o   = c >> 2;
                int off = (c & 3) << 4;
                copy16_g2l(wq + (size_t)(tile_o + o) * IN_F + ks * BK + off,
                           &Bs[buf][o * LDK + off]);
            }
        } else {
#pragma unroll
            for (int it = 0; it < 4; ++it) {
                int idx = it * 256 + tid;
                int o   = idx >> 3;
                int wrd = idx & 7;
                unsigned p = (unsigned)packed_w[(size_t)(tile_o + o) * (IN_F / 8) + ks * 8 + wrd];
                v2i w; w[0] = (int)nib_lo(p); w[1] = (int)nib_hi(p);
                *(v2i*)(&Bs[buf][o * LDK + wrd * 8]) = w;
            }
        }
    };

    stage(0, 0);
    wait_stage();
    __syncthreads();

#pragma unroll 1
    for (int ks = 0; ks < KSTEPS; ++ks) {
        const int cur = ks & 1;
        if (ks + 1 < KSTEPS) stage(ks + 1, cur ^ 1);

        const signed char* Ab = &As[cur][0];
        const signed char* Bb = &Bs[cur][0];

        v8i a[4], b[2];
#pragma unroll
        for (int ti = 0; ti < 4; ++ti)
            a[ti] = ld_fragA(Ab + (wm * 64 + ti * 16 + rlo) * LDK + khA);
#pragma unroll
        for (int tj = 0; tj < 2; ++tj)
            b[tj] = ld_fragB(Bb + (wn * 32 + tj * 16 + rlo) * LDK + khB);

#pragma unroll
        for (int ti = 0; ti < 4; ++ti)
#pragma unroll
            for (int tj = 0; tj < 2; ++tj)
                acc[ti][tj] = __builtin_amdgcn_wmma_i32_16x16x64_iu8(
                     true,  a[ti],
                     false, b[tj],
                    acc[ti][tj],
                     false,  false);

        wait_stage();
        __syncthreads();
    }

    const float act_s = act_scale[0];
#pragma unroll
    for (int tj = 0; tj < 2; ++tj) {
        const int o  = tile_o + wn * 32 + tj * 16 + rlo;
        const float so = act_s / scale[o];
        const float zp = zero_point[o];
        const float bz = bias[o];
#pragma unroll
        for (int ti = 0; ti < 4; ++ti) {
            const int nb = tile_m + wm * 64 + ti * 16 + ((lane >> 4) << 3);
#pragma unroll
            for (int r = 0; r < 8; ++r) {
                const int n = nb + r;
                const float rs = (float)rowsum[n];
                out[(size_t)n * OUT_F + o] = so * ((float)acc[ti][tj][r] + zp * rs) + bz;
            }
        }
    }
}

extern "C" void kernel_launch(void* const* d_in, const int* in_sizes, int n_in,
                              void* d_out, int out_size, void* d_ws, size_t ws_size,
                              hipStream_t stream) {
    const float* x          = (const float*)d_in[0];
    const int*   packed_w   = (const int*)d_in[1];
    const float* scale      = (const float*)d_in[2];
    const float* zero_point = (const float*)d_in[3];
    const float* bias       = (const float*)d_in[4];
    const float* act_scale  = (const float*)d_in[5];
    float*       out        = (float*)d_out;

    const size_t XQ_BYTES = (size_t)N_TOKENS * IN_F;
    const size_t RS_BYTES = (size_t)N_TOKENS * 4;
    const size_t WQ_BYTES = (size_t)OUT_F * IN_F;

    signed char* xq     = (signed char*)d_ws;
    int*         rowsum = (int*)((char*)d_ws + XQ_BYTES);
    signed char* wq     = (signed char*)d_ws + XQ_BYTES + RS_BYTES;

    quant_rows_kernel<<<N_TOKENS, 256, 0, stream>>>(x, act_scale, xq, rowsum);

    dim3 grid(N_TOKENS / BM, OUT_F / BN);
    if (ws_size >= XQ_BYTES + RS_BYTES + WQ_BYTES) {
        unpack_w_kernel<<<(OUT_F * IN_F / 16) / 256, 256, 0, stream>>>(packed_w, wq);
        w4a8_gemm_kernel<true><<<grid, 256, 0, stream>>>(
            xq, wq, packed_w, rowsum, scale, zero_point, bias, act_scale, out);
    } else {
        w4a8_gemm_kernel<false><<<grid, 256, 0, stream>>>(
            xq, nullptr, packed_w, rowsum, scale, zero_point, bias, act_scale, out);
    }
}
